// FlashAttention2_1005022347992
// MI455X (gfx1250) — matrix operand order repaired and checked by reading, not compiled, not run on hardware
//
#include <hip/hip_runtime.h>

// Problem constants (match reference)
#define B_    2
#define S_    2048
#define H_    1024
#define NH_   16
#define HD_   64

// Toggle for the Tensor Data Mover path (inline asm, per CDNA5 ISA)
#define USE_TDM 1

typedef unsigned int u32;
typedef __attribute__((ext_vector_type(8)))  float  v8f;
typedef __attribute__((ext_vector_type(4)))  float  v4f;
typedef __attribute__((ext_vector_type(16))) __bf16 v16bf;
typedef __attribute__((ext_vector_type(8)))  __bf16 v8bf;
typedef __attribute__((ext_vector_type(4)))  u32    v4u;
typedef __attribute__((ext_vector_type(8)))  u32    v8u;

static __device__ __forceinline__ v16bf bf_join(v8bf lo, v8bf hi) {
  return __builtin_shufflevector(lo, hi, 0,1,2,3,4,5,6,7,8,9,10,11,12,13,14,15);
}

static __device__ __forceinline__ v8f wmma_bf16(v16bf a, v16bf b, v8f c) {
  // (neg_a, A, neg_b, B, c_mod, C, reuse_a, reuse_b)
  return __builtin_amdgcn_wmma_f32_16x16x32_bf16(false, a, false, b, (short)0, c,
                                                 false, false);
}

#if USE_TDM
// Tensor DMA Descriptor group 0 (128b): count=1, lds_addr, global_addr, type=2
static __device__ __forceinline__ v4u tdm_g0(u32 lds_off, const void* gaddr) {
  unsigned long long ga = (unsigned long long)gaddr;
  v4u g0;
  g0[0] = 1u;                               // [1:0] count=1; gather off; rest 0
  g0[1] = lds_off;                          // [63:32] lds_addr (bytes)
  g0[2] = (u32)ga;                          // [95:64] global_addr lo
  g0[3] = ((u32)(ga >> 32) & 0x01FFFFFFu)   // [120:96] global_addr hi (57-bit)
          | (2u << 30);                     // [127:126] type=2 ("image")
  return g0;
}
// Group 1 (256b): 2D tile, bf16 (data_size=1), no multicast/pad/iterate
static __device__ __forceinline__ v8u tdm_g1(u32 dim0, u32 dim1, u32 tile0,
                                             u32 tile1, u32 stride0) {
  v8u g1 = {};
  g1[0] = (1u << 16);                              // [17:16] data_size = 1 (2B)
  g1[1] = (dim0 & 0xFFFFu) << 16;                  // [79:48]  tensor_dim0 lo16
  g1[2] = ((dim0 >> 16) & 0xFFFFu)                 //          tensor_dim0 hi16
          | ((dim1 & 0xFFFFu) << 16);              // [111:80] tensor_dim1 lo16
  g1[3] = ((dim1 >> 16) & 0xFFFFu)                 //          tensor_dim1 hi16
          | ((tile0 & 0xFFFFu) << 16);             // [127:112] tile_dim0
  g1[4] = (tile1 & 0xFFFFu);                       // [143:128] tile_dim1
  g1[5] = stride0;                                 // [207:160] tensor_dim0_stride lo
  g1[6] = 0;                                       // stride0 hi, stride1 lo
  g1[7] = 0;                                       // stride1 hi
  return g1;
}
static __device__ __forceinline__ void tdm_load(v4u g0, v8u g1) {
  asm volatile("tensor_load_to_lds %0, %1" :: "s"(g0), "s"(g1) : "memory");
}
#endif

// ---------------------------------------------------------------------------
// GEMM: out = A(fp32 [M x H]) * W^T(fp32 [H_out x H]) + bias
// MODE 0: write bf16, head-split layout [b, head, s, d]  (for Q/K/V)
// MODE 1: write fp32 row-major [M x H]                   (final output)
// Block: 256 threads, 128x128 output tile; wave -> 32x64 quadrant (8 WMMA/k).
// ---------------------------------------------------------------------------
template <int MODE>
__global__ __launch_bounds__(256) void proj_gemm(
    const float* __restrict__ A, const float* __restrict__ W,
    const float* __restrict__ bias, __bf16* __restrict__ outb,
    float* __restrict__ outf) {
  __shared__ __align__(32) __bf16 As[128][32];   // 8 KB
  __shared__ __align__(32) __bf16 Bs[128][32];   // 8 KB

  const int tid  = threadIdx.x;
  const int lane = tid & 31;
  const int wave = tid >> 5;
  const int wr = wave >> 1, wc = wave & 1;
  const int m0 = blockIdx.y * 128;
  const int n0 = blockIdx.x * 128;

  v8f acc[2][4] = {};

  // staging indices: 128 rows x two 16-col segments per tile
  const int sr = tid >> 1, sseg = (tid & 1) * 16;
  // fragment gather indices (contiguous k order k = 16*(lane/16) + s, same for A and B)
  const int mlane = lane & 15;
  const int kA0 = (lane & 16) ? 16 : 0;
  const int kB0 = (lane & 16) ? 16 : 0;
  const int mloc = wr * 32 + mlane;
  const int nloc = wc * 64 + mlane;

  for (int kb = 0; kb < H_; kb += 32) {
    // stage A tile (fp32 -> bf16)
    {
      const v4f* ap = (const v4f*)(A + (size_t)(m0 + sr) * H_ + kb + sseg);
      v4f x0 = ap[0], x1 = ap[1], x2 = ap[2], x3 = ap[3];
      v8bf lo, hi;
      for (int i = 0; i < 4; ++i) {
        lo[i] = (__bf16)x0[i]; lo[i + 4] = (__bf16)x1[i];
        hi[i] = (__bf16)x2[i]; hi[i + 4] = (__bf16)x3[i];
      }
      *(v8bf*)&As[sr][sseg]     = lo;
      *(v8bf*)&As[sr][sseg + 8] = hi;
    }
    // stage B tile: rows of W are contiguous in k (B = W^T columns)
    {
      const v4f* wp = (const v4f*)(W + (size_t)(n0 + sr) * H_ + kb + sseg);
      v4f x0 = wp[0], x1 = wp[1], x2 = wp[2], x3 = wp[3];
      v8bf lo, hi;
      for (int i = 0; i < 4; ++i) {
        lo[i] = (__bf16)x0[i]; lo[i + 4] = (__bf16)x1[i];
        hi[i] = (__bf16)x2[i]; hi[i + 4] = (__bf16)x3[i];
      }
      *(v8bf*)&Bs[sr][sseg]     = lo;
      *(v8bf*)&Bs[sr][sseg + 8] = hi;
    }
    __syncthreads();

    v16bf aF[2], bF[4];
    for (int mt = 0; mt < 2; ++mt) {
      const __bf16* p = &As[mloc + mt * 16][0];
      aF[mt] = bf_join(*(const v8bf*)(p + kA0), *(const v8bf*)(p + kA0 + 8));
    }
    for (int nt = 0; nt < 4; ++nt)
      bF[nt] = *(const v16bf*)&Bs[nloc + nt * 16][kB0];

    for (int mt = 0; mt < 2; ++mt)
      for (int nt = 0; nt < 4; ++nt)
        acc[mt][nt] = wmma_bf16(aF[mt], bF[nt], acc[mt][nt]);
    __syncthreads();
  }

  // epilogue: C/D layout -> (row, col), add bias
  const int halfsel = (lane & 16) ? 8 : 0;
  for (int mt = 0; mt < 2; ++mt)
    for (int nt = 0; nt < 4; ++nt)
      for (int r = 0; r < 8; ++r) {
        int row = m0 + wr * 32 + mt * 16 + r + halfsel;
        int col = n0 + wc * 64 + nt * 16 + mlane;
        float v = acc[mt][nt][r] + bias[col];
        if (MODE == 0) {
          int b = row >> 11, s = row & (S_ - 1);    // S_ = 2048
          int head = col >> 6, d = col & 63;        // HD_ = 64
          outb[((size_t)(b * NH_ + head) * S_ + s) * HD_ + d] = (__bf16)v;
        } else {
          outf[(size_t)row * H_ + col] = v;
        }
      }
}

// ---------------------------------------------------------------------------
// Flash attention: one block per (b, head, 128 query rows); 8 waves, each wave
// owns a 16-row query tile. Online softmax in fp32, QK^T and P*V on WMMA bf16.
// K tile staged by the Tensor Data Mover; V staged manually (transposed).
// ---------------------------------------------------------------------------
__global__ __launch_bounds__(256) void flash_attn(
    const __bf16* __restrict__ Q, const __bf16* __restrict__ K,
    const __bf16* __restrict__ V, float* __restrict__ ctx) {
  __shared__ __align__(32) __bf16 Ks[32][64];      // 4 KB  (keys x d)
  __shared__ __align__(32) __bf16 VTs[64][32];     // 4 KB  (d x keys, transposed)
  __shared__ __align__(32) __bf16 Ps[8][16][32];   // 8 KB  per-wave P staging

  const int tid  = threadIdx.x;
  const int lane = tid & 31;
  const int wave = tid >> 5;
  const int qblk = blockIdx.x & 15;   // S_/128 = 16 query blocks
  const int bh   = blockIdx.x >> 4;
  const size_t base = (size_t)bh * S_ * HD_;

  const int q0 = qblk * 128 + wave * 16;
  const int mlane = lane & 15;
  const int halfsel = (lane & 16) ? 8 : 0;
  const int kA0 = (lane & 16) ? 16 : 0;
  const int kB0 = (lane & 16) ? 16 : 0;

  // Q fragments for both 32-wide k-chunks of head_dim=64, kept in registers
  v16bf qF[2];
  {
    const __bf16* qrow = Q + base + (size_t)(q0 + mlane) * HD_;
    for (int c = 0; c < 2; ++c) {
      int k0 = c * 32 + kA0;
      qF[c] = bf_join(*(const v8bf*)(qrow + k0), *(const v8bf*)(qrow + k0 + 8));
    }
  }

  float mrow[8], lrow[8];
  v8f acc[4] = {};   // 16q x 64d context accumulator (four 16x16 tiles)
  for (int r = 0; r < 8; ++r) { mrow[r] = -1e30f; lrow[r] = 0.f; }

  const int str = tid >> 3;        // 0..31 key row
  const int stc = (tid & 7) * 8;   // 0..56 d segment

#if USE_TDM
  const u32 ksLds = (u32)(uintptr_t)&Ks[0][0];   // LDS byte offset (low 32 bits)
#endif

  for (int kb = 0; kb < S_; kb += 32) {
#if USE_TDM
    // K tile via Tensor Data Mover: 2D tile 64(d) x 32(keys), bf16, row-major.
    if (wave == 0) {
      v4u g0 = tdm_g0(ksLds, K + base + (size_t)kb * HD_);
      v8u g1 = tdm_g1(/*dim0=*/HD_, /*dim1=*/(u32)(S_ - kb),
                      /*tile0=*/HD_, /*tile1=*/32, /*stride0=*/HD_);
      tdm_load(g0, g1);
    }
#else
    *(v8bf*)&Ks[str][stc] = *(const v8bf*)(K + base + (size_t)(kb + str) * HD_ + stc);
#endif
    // V tile staged transposed (PV B-fragment wants contiguous k per d)
    v8bf vv = *(const v8bf*)(V + base + (size_t)(kb + str) * HD_ + stc);
    for (int i = 0; i < 8; ++i) VTs[stc + i][str] = vv[i];

    // prefetch next key block while we compute on this one
    if (kb + 32 < S_) {
      __builtin_prefetch(K + base + (size_t)(kb + 32 + str) * HD_ + stc, 0, 0);
      __builtin_prefetch(V + base + (size_t)(kb + 32 + str) * HD_ + stc, 0, 0);
    }
#if USE_TDM
    __builtin_amdgcn_s_wait_tensorcnt(0);   // no-op for waves with TENSORcnt==0
#endif
    __syncthreads();

    // scores = Q * K^T for two 16-key subtiles (K-dim = 64 -> 2 WMMAs each)
    v8f s0 = {}, s1 = {};
    for (int c = 0; c < 2; ++c) {
      int kk = c * 32 + kB0;
      v16bf b0 = *(const v16bf*)&Ks[mlane][kk];
      v16bf b1 = *(const v16bf*)&Ks[16 + mlane][kk];
      s0 = wmma_bf16(qF[c], b0, s0);
      s1 = wmma_bf16(qF[c], b1, s1);
    }
    s0 *= 0.125f;  // 1/sqrt(64)
    s1 *= 0.125f;

    // online softmax: row reductions across the 16-lane half via xor shuffles
    float alpha[8];
    for (int r = 0; r < 8; ++r) {
      float v = fmaxf(s0[r], s1[r]);
      for (int m = 1; m < 16; m <<= 1) v = fmaxf(v, __shfl_xor(v, m, 32));
      float mn = fmaxf(mrow[r], v);
      alpha[r] = __expf(mrow[r] - mn);
      float p0 = __expf(s0[r] - mn);
      float p1 = __expf(s1[r] - mn);
      float rs = p0 + p1;
      for (int m = 1; m < 16; m <<= 1) rs += __shfl_xor(rs, m, 32);
      lrow[r] = lrow[r] * alpha[r] + rs;
      mrow[r] = mn;
      s0[r] = p0; s1[r] = p1;
    }
    for (int t = 0; t < 4; ++t)
      for (int r = 0; r < 8; ++r) acc[t][r] *= alpha[r];

    // C-layout -> A-layout for P via per-wave LDS staging
    for (int r = 0; r < 8; ++r) {
      int row = r + halfsel;
      Ps[wave][row][mlane]      = (__bf16)s0[r];
      Ps[wave][row][mlane + 16] = (__bf16)s1[r];
    }
    const __bf16* prow = &Ps[wave][mlane][0];
    v16bf pF = bf_join(*(const v8bf*)(prow + kA0), *(const v8bf*)(prow + kA0 + 8));

    // acc += P(16x32) * V(32x64): B fragments contiguous thanks to V transpose
    for (int t = 0; t < 4; ++t) {
      v16bf bv = *(const v16bf*)&VTs[t * 16 + mlane][kB0];
      acc[t] = wmma_bf16(pF, bv, acc[t]);
    }
    __syncthreads();
  }

  // normalize and write fp32 context, head-merged [b, s, h]
  const int b  = bh >> 4;   // NH_ = 16
  const int hd = bh & 15;
  for (int r = 0; r < 8; ++r) {
    float inv = 1.0f / lrow[r];
    int srow = q0 + r + halfsel;
    float* orow = ctx + (size_t)(b * S_ + srow) * H_ + hd * HD_;
    for (int t = 0; t < 4; ++t) orow[t * 16 + mlane] = acc[t][r] * inv;
  }
}

// ---------------------------------------------------------------------------
extern "C" void kernel_launch(void* const* d_in, const int* in_sizes, int n_in,
                              void* d_out, int out_size, void* d_ws, size_t ws_size,
                              hipStream_t stream) {
  (void)in_sizes; (void)n_in; (void)out_size; (void)ws_size;
  const float* hidden = (const float*)d_in[0];
  const float* Wq = (const float*)d_in[1]; const float* bq = (const float*)d_in[2];
  const float* Wk = (const float*)d_in[3]; const float* bk = (const float*)d_in[4];
  const float* Wv = (const float*)d_in[5]; const float* bv = (const float*)d_in[6];
  const float* Wo = (const float*)d_in[7]; const float* bo = (const float*)d_in[8];

  char* ws = (char*)d_ws;
  const size_t qkvBytes = (size_t)B_ * S_ * H_ * sizeof(__bf16);  // 8 MB each
  __bf16* Qbf = (__bf16*)(ws);
  __bf16* Kbf = (__bf16*)(ws + qkvBytes);
  __bf16* Vbf = (__bf16*)(ws + 2 * qkvBytes);
  float*  ctx = (float*)(ws + 3 * qkvBytes);                      // 16 MB fp32

  dim3 pgrid(H_ / 128, (B_ * S_) / 128);  // (8, 32)
  proj_gemm<0><<<pgrid, 256, 0, stream>>>(hidden, Wq, bq, Qbf, nullptr);
  proj_gemm<0><<<pgrid, 256, 0, stream>>>(hidden, Wk, bk, Kbf, nullptr);
  proj_gemm<0><<<pgrid, 256, 0, stream>>>(hidden, Wv, bv, Vbf, nullptr);

  flash_attn<<<B_ * NH_ * (S_ / 128), 256, 0, stream>>>(Qbf, Kbf, Vbf, ctx);

  proj_gemm<1><<<pgrid, 256, 0, stream>>>(ctx, Wo, bo, nullptr, (float*)d_out);
}
